// MultiQueryAttentionLayer_63453846831363
// MI455X (gfx1250) — compile-verified, matrix operand order checked by reading, not run on hardware
//
#include <hip/hip_runtime.h>
#include <hip/hip_bf16.h>

// ---------------------------------------------------------------------------
// MQA layer for MI455X (gfx1250, wave32).  All GEMMs via
// v_wmma_f32_16x16x32_bf16 (f32 accumulate).  GEMM A-panels staged with the
// Tensor Data Mover (tensor_load_to_lds + s_wait_tensorcnt), W panels staged
// transposed through LDS.  Fused masked-softmax attention with attn
// materialization + P@V in LDS.
//
// WMMA 16x16x32 bf16 lane layouts (CDNA5 ISA 7.12.2), hh=lane>>4, r=lane&15:
//   A (16x32): a[j] = A[r][(j>>3)*16 + hh*8 + (j&7)]
//   B (32x16): b[j] = B[(j>>3)*16 + hh*8 + (j&7)][r]
//   C (16x16): c[v] = C[v + 8*hh][r]
// ---------------------------------------------------------------------------

typedef __attribute__((ext_vector_type(16))) __bf16 v16bf;
typedef __attribute__((ext_vector_type(8)))  float  v8f;
typedef __attribute__((ext_vector_type(4)))  unsigned int v4u;
typedef __attribute__((ext_vector_type(8)))  int    v8i;
typedef __attribute__((ext_vector_type(4)))  int    v4i;

#define BB   4
#define SS   2048
#define HIDN 1024
#define HH   16
#define DD   64

struct bf16x8 { __bf16 v[8]; };   // 16-byte vector chunk

static __device__ inline v8f wmma_bf16(v16bf a, v16bf b, v8f c) {
  return __builtin_amdgcn_wmma_f32_16x16x32_bf16(
      /*neg_a=*/false, a, /*neg_b=*/false, b,
      /*c_mod=*/(short)0, c, /*reuse_a=*/false, /*reuse_b=*/false);
}

// Load 8 contiguous bf16 (16B) into frag elements [lo, lo+8)
static __device__ inline void ld8(v16bf& f, int lo, const __bf16* p) {
  bf16x8 t = *(const bf16x8*)p;
#pragma unroll
  for (int j = 0; j < 8; ++j) f[lo + j] = t.v[j];
}

// ---------------------------------------------------------------------------
// TDM: DMA a 2-D tile (tileD0 x tileD1 elements of 2 bytes) from a row-major
// tensor (row length strideD0 elements) into LDS at byte offset ldsOff.
// D# packing per CDNA5 ISA.
// ---------------------------------------------------------------------------
static __device__ inline void tdm_load_2d_bf16(
    unsigned ldsOff, const void* gptr,
    unsigned tensorD0, unsigned tensorD1, unsigned strideD0,
    unsigned tileD0, unsigned tileD1) {
  unsigned long long ga = (unsigned long long)gptr;
  v4u g0;
  g0[0] = 1u;                                       // count=1, no gather
  g0[1] = ldsOff;                                   // lds_addr
  g0[2] = (unsigned)(ga & 0xffffffffu);             // global_addr[31:0]
  g0[3] = (unsigned)((ga >> 32) & 0x01ffffffu)      // global_addr[56:32]
          | (2u << 30);                             // type = 2 ("image")
  v8i g1;
  g1[0] = (int)(1u << 16);                          // data_size=1 (2 bytes)
  g1[1] = (int)((tensorD0 & 0xffffu) << 16);        // tensor_dim0[15:0]
  g1[2] = (int)((tensorD0 >> 16) | ((tensorD1 & 0xffffu) << 16));
  g1[3] = (int)((tensorD1 >> 16) | (tileD0 << 16)); // tile_dim0
  g1[4] = (int)(tileD1 & 0xffffu);                  // tile_dim1 (tile_dim2=0)
  g1[5] = (int)strideD0;                            // tensor_dim0_stride[31:0]
  g1[6] = 0;                                        // stride0[47:32], stride1 lo
  g1[7] = 0;                                        // stride1 hi
  v4i z4 = {0, 0, 0, 0};
#if defined(__clang_major__) && (__clang_major__ >= 23)
  v8i z8 = {0, 0, 0, 0, 0, 0, 0, 0};
  __builtin_amdgcn_tensor_load_to_lds(g0, g1, z4, z4, z8, 0);
#else
  __builtin_amdgcn_tensor_load_to_lds(g0, g1, z4, z4, 0);
#endif
}

// ---------------------------------------------------------------------------
// f32 -> bf16 pack (grid-stride)
// ---------------------------------------------------------------------------
__global__ void pack_bf16_k(const float* __restrict__ in,
                            __bf16* __restrict__ out, int n) {
  int i = blockIdx.x * blockDim.x + threadIdx.x;
  int stride = gridDim.x * blockDim.x;
  for (; i < n; i += stride) out[i] = (__bf16)in[i];
}

// ---------------------------------------------------------------------------
// C[M,N] = A[M,K] @ W[K,N] + bias[N]   (bf16 in, f32 accum)
// 256 threads, 64x64 macro-tile, 8 waves x two 16x16 C tiles.
// LDS: A panel row-major [64][32] (filled by TDM); W panel transposed
// Wt[n][k] (pitch 40 halves, 16B aligned rows).
// transposeV: store bf16 result per-batch transposed [B][N][2048] (for V/Vi).
// ---------------------------------------------------------------------------
#define WT_PITCH 40

__global__ __launch_bounds__(256) void wmma_gemm_bias(
    const __bf16* __restrict__ A, const __bf16* __restrict__ W,
    const float* __restrict__ bias,
    float* __restrict__ Cf, __bf16* __restrict__ Cbf,
    int M, int N, int K, int transposeV) {
  __shared__ __bf16 As[64 * 32];
  __shared__ __bf16 Wt[64 * WT_PITCH];

  const int tid  = threadIdx.x;
  const int wave = tid >> 5;
  const int lane = tid & 31;
  const int hh   = lane >> 4;
  const int r    = lane & 15;

  const int tilesN = N >> 6;
  const int tm = blockIdx.x / tilesN;
  const int tn = blockIdx.x - tm * tilesN;
  const int m0 = tm * 64;
  const int n0 = tn * 64;

  const int ms0 = wave >> 2;       // m-subtiles ms0 and ms0+2
  const int ns  = wave & 3;        // n-subtile
  const int n   = n0 + ns * 16 + r;

  // W staging coordinates: 32 k-rows x 8 chunks of 8 bf16
  const int wk = tid >> 3, wn8 = (tid & 7) * 8;
  // LDS byte offset of As (flat addr truncated to 32 bits = LDS offset)
  const unsigned asOff = (unsigned)(unsigned long long)(const void*)As;

  const float bval = bias[n];
  v8f c0, c1;
#pragma unroll
  for (int v = 0; v < 8; ++v) { c0[v] = bval; c1[v] = bval; }

  for (int kk = 0; kk < K; kk += 32) {
    __syncthreads();   // previous iteration's fragment reads complete
    // ---- stage A panel via Tensor Data Mover (wave 0 issues the DMA)
    if (tid < 32) {
      tdm_load_2d_bf16(asOff, A + (size_t)m0 * K + kk,
                       (unsigned)K, (unsigned)M, (unsigned)K,
                       /*tileD0=*/32u, /*tileD1=*/64u);
    }
    // ---- stage W panel transposed (global b128 -> 8x ds b16)
    {
      const __bf16* wp = W + (size_t)(kk + wk) * N + n0 + wn8;
      bf16x8 t = *(const bf16x8*)wp;
#pragma unroll
      for (int j = 0; j < 8; ++j) Wt[(wn8 + j) * WT_PITCH + wk] = t.v[j];
      if (kk + 32 < K) __builtin_prefetch(wp + (size_t)32 * N, 0, 3);
    }
    if (tid < 32) __builtin_amdgcn_s_wait_tensorcnt(0);
    __syncthreads();   // TDM + W staging visible to all waves

    // ---- fragments from LDS (contiguous ds_load_b128)
    v16bf bf_;
    ld8(bf_, 0, Wt + (ns * 16 + r) * WT_PITCH + hh * 8);
    ld8(bf_, 8, Wt + (ns * 16 + r) * WT_PITCH + 16 + hh * 8);
    v16bf a0, a1;
    ld8(a0, 0, As + (ms0 * 16 + r) * 32 + hh * 8);
    ld8(a0, 8, As + (ms0 * 16 + r) * 32 + 16 + hh * 8);
    ld8(a1, 0, As + ((ms0 + 2) * 16 + r) * 32 + hh * 8);
    ld8(a1, 8, As + ((ms0 + 2) * 16 + r) * 32 + 16 + hh * 8);

    c0 = wmma_bf16(a0, bf_, c0);
    c1 = wmma_bf16(a1, bf_, c1);
  }

  // ---- store both C tiles
#pragma unroll
  for (int t = 0; t < 2; ++t) {
    const v8f cc = t ? c1 : c0;
    const int ms = ms0 + t * 2;
#pragma unroll
    for (int v = 0; v < 8; ++v) {
      const int m = m0 + ms * 16 + v + 8 * hh;
      if (transposeV) {
        // per-batch transpose: m = b*2048 + s -> Ct[b][n][s]
        const int bb = m >> 11, s = m & 2047;
        Cbf[((size_t)bb * N + n) * SS + s] = (__bf16)cc[v];
      } else {
        const size_t idx = (size_t)m * N + n;
        if (Cf)  Cf[idx]  = cc[v];
        if (Cbf) Cbf[idx] = (__bf16)cc[v];
      }
    }
  }
}

// ---------------------------------------------------------------------------
// Fused attention phase: energy (Q@K^T/8, masked) -> softmax -> optional
// attn store -> accumulate P@V into xacc (LDS).  e: [16][SS] f32 in LDS.
// Vp is TRANSPOSED: [B][DD][SS].
// ---------------------------------------------------------------------------
static __device__ inline void attn_phase(
    const __bf16* __restrict__ Qp, const __bf16* __restrict__ Kp,
    const __bf16* __restrict__ Vp, const int* __restrict__ maskRow,
    float* e, float* xacc, float* __restrict__ attn_out,  // attn_out may be null
    int b, int h, int q0) {
  const int tid  = threadIdx.x;
  const int wave = tid >> 5;
  const int lane = tid & 31;
  const int hh   = lane >> 4;
  const int r    = lane & 15;

  // ---- scores: Q tile (16x64) vs all 2048 keys; all frag loads contiguous
  const __bf16* qrow = Qp + (size_t)(b * SS + q0 + r) * HIDN + h * DD;
  v16bf A0, A1;
  ld8(A0, 0, qrow + hh * 8);        ld8(A0, 8, qrow + 16 + hh * 8);
  ld8(A1, 0, qrow + 32 + hh * 8);   ld8(A1, 8, qrow + 48 + hh * 8);

  for (int st = wave; st < (SS / 16); st += 8) {
    const int s0 = st * 16;
    const __bf16* krow = Kp + (size_t)(b * SS + s0 + r) * DD;
    v16bf B0, B1;
    ld8(B0, 0, krow + hh * 8);       ld8(B0, 8, krow + 16 + hh * 8);
    ld8(B1, 0, krow + 32 + hh * 8);  ld8(B1, 8, krow + 48 + hh * 8);
    v8f c = {};
    c = wmma_bf16(A0, B0, c);
    c = wmma_bf16(A1, B1, c);
    const int keep = maskRow[s0 + r];
#pragma unroll
    for (int v = 0; v < 8; ++v) {
      float val = keep ? c[v] * 0.125f : -1.0e10f;
      e[(v + 8 * hh) * SS + s0 + r] = val;
    }
  }
  __syncthreads();

  // ---- row softmax: 16 threads per row, shuffle reduce within 16 lanes
  {
    const int row = tid >> 4;
    const int sub = tid & 15;
    float* erow = e + row * SS;
    float m = -3.0e38f;
    for (int t = sub; t < SS; t += 16) m = fmaxf(m, erow[t]);
#pragma unroll
    for (int o = 8; o >= 1; o >>= 1) m = fmaxf(m, __shfl_xor(m, o, 16));
    float sum = 0.f;
    for (int t = sub; t < SS; t += 16) {
      float p = __expf(erow[t] - m);
      erow[t] = p;
      sum += p;
    }
#pragma unroll
    for (int o = 8; o >= 1; o >>= 1) sum += __shfl_xor(sum, o, 16);
    const float inv = 1.0f / sum;
    if (attn_out) {
      float* orow = attn_out +
          ((size_t)(b * HH + h) * SS + q0 + row) * (size_t)SS;
      for (int t = sub; t < SS; t += 16) {
        float p = erow[t] * inv;
        erow[t] = p;
        orow[t] = p;
      }
    } else {
      for (int t = sub; t < SS; t += 16) erow[t] *= inv;
    }
  }
  __syncthreads();

  // ---- x += P[16,2048] @ V[2048,64]; waves: kw = k-half, dt = 16-col d tile
  {
    const int dt = wave & 3;
    const int kw = wave >> 2;
    v8f acc = {};
    const __bf16* vrowBase = Vp + ((size_t)b * DD + dt * 16 + r) * SS;
    for (int ch = kw * 32; ch < kw * 32 + 32; ++ch) {
      const int c0v = ch * 32;
      v16bf Af, Bf;
      const float* prow = e + r * SS + c0v;
#pragma unroll
      for (int j = 0; j < 8; ++j) {
        Af[j]     = (__bf16)prow[hh * 8 + j];
        Af[8 + j] = (__bf16)prow[16 + hh * 8 + j];
      }
      const __bf16* vrow = vrowBase + c0v;     // transposed V: contiguous in k
      ld8(Bf, 0, vrow + hh * 8);
      ld8(Bf, 8, vrow + 16 + hh * 8);
      acc = wmma_bf16(Af, Bf, acc);
    }
#pragma unroll
    for (int v = 0; v < 8; ++v)
      atomicAdd(&xacc[(v + 8 * hh) * DD + dt * 16 + r], acc[v]);
  }
  __syncthreads();
}

// ---------------------------------------------------------------------------
// One block per (b, h, 16-query block).  Dynamic LDS:
//   e    : 16*2048 f32  (128 KB)
//   xacc : 16*64   f32  (4 KB)
// ---------------------------------------------------------------------------
__global__ __launch_bounds__(256) void attn_kernel(
    const __bf16* __restrict__ Qb,  const __bf16* __restrict__ Qib,
    const __bf16* __restrict__ Kb,  const __bf16* __restrict__ Kib,
    const __bf16* __restrict__ Vb,  const __bf16* __restrict__ Vib,
    const int* __restrict__ mask, float* __restrict__ attn_out,
    __bf16* __restrict__ Xb) {
  extern __shared__ float smem_f[];
  float* e    = smem_f;
  float* xacc = smem_f + 16 * SS;

  const int b  = blockIdx.z;
  const int h  = blockIdx.y;
  const int q0 = blockIdx.x * 16;
  const int* maskRow = mask + b * SS;

  for (int i = threadIdx.x; i < 16 * DD; i += 256) xacc[i] = 0.f;
  __syncthreads();

  attn_phase(Qb,  Kb,  Vb,  maskRow, e, xacc, attn_out, b, h, q0);  // main
  attn_phase(Qib, Kib, Vib, maskRow, e, xacc, nullptr,  b, h, q0);  // iter

  for (int i = threadIdx.x; i < 16 * DD; i += 256) {
    int row = i >> 6, d = i & 63;
    Xb[(size_t)(b * SS + q0 + row) * HIDN + h * DD + d] = (__bf16)xacc[i];
  }
}

// ---------------------------------------------------------------------------
extern "C" void kernel_launch(void* const* d_in, const int* in_sizes, int n_in,
                              void* d_out, int out_size, void* d_ws,
                              size_t ws_size, hipStream_t stream) {
  const float* query = (const float*)d_in[0];
  const float* key_  = (const float*)d_in[1];
  const float* value = (const float*)d_in[2];
  const float* iter  = (const float*)d_in[3];
  const int*   mask  = (const int*)d_in[4];
  const float* Wq  = (const float*)d_in[5];  const float* bq  = (const float*)d_in[6];
  const float* Wk  = (const float*)d_in[7];  const float* bk  = (const float*)d_in[8];
  const float* Wv  = (const float*)d_in[9];  const float* bv  = (const float*)d_in[10];
  const float* Wiq = (const float*)d_in[11]; const float* biq = (const float*)d_in[12];
  const float* Wik = (const float*)d_in[13]; const float* bik = (const float*)d_in[14];
  const float* Wiv = (const float*)d_in[15]; const float* biv = (const float*)d_in[16];
  const float* Wo  = (const float*)d_in[17]; const float* bo  = (const float*)d_in[18];

  float* out_x    = (float*)d_out;                         // [B,S,HID]
  float* out_attn = out_x + (size_t)BB * SS * HIDN;        // [B,H,S,S]

  const int M   = BB * SS;          // 8192
  const int ACT = M * HIDN;         // 8,388,608
  const int KVN = M * DD;           // 524,288

  char* wsp = (char*)d_ws;
  auto alloc = [&](size_t bytes) { char* p = wsp; wsp += bytes; return p; };
  __bf16* qbf   = (__bf16*)alloc((size_t)ACT * 2);
  __bf16* kbf   = (__bf16*)alloc((size_t)ACT * 2);
  __bf16* vbf   = (__bf16*)alloc((size_t)ACT * 2);
  __bf16* ibf   = (__bf16*)alloc((size_t)ACT * 2);
  __bf16* wqbf  = (__bf16*)alloc((size_t)HIDN * HIDN * 2);
  __bf16* wkbf  = (__bf16*)alloc((size_t)HIDN * DD * 2);
  __bf16* wvbf  = (__bf16*)alloc((size_t)HIDN * DD * 2);
  __bf16* wiqbf = (__bf16*)alloc((size_t)HIDN * HIDN * 2);
  __bf16* wikbf = (__bf16*)alloc((size_t)HIDN * DD * 2);
  __bf16* wivbf = (__bf16*)alloc((size_t)HIDN * DD * 2);
  __bf16* wobf  = (__bf16*)alloc((size_t)HIDN * HIDN * 2);
  __bf16* Qbf   = (__bf16*)alloc((size_t)ACT * 2);
  __bf16* Qibf  = (__bf16*)alloc((size_t)ACT * 2);
  __bf16* Kbf   = (__bf16*)alloc((size_t)KVN * 2);
  __bf16* Vbf   = (__bf16*)alloc((size_t)KVN * 2);   // [B][DD][SS]
  __bf16* Kibf  = (__bf16*)alloc((size_t)KVN * 2);
  __bf16* Vibf  = (__bf16*)alloc((size_t)KVN * 2);   // [B][DD][SS]
  __bf16* Xbf   = (__bf16*)alloc((size_t)ACT * 2);

  // ---- pack f32 -> bf16
  auto pack = [&](const float* src, __bf16* dst, int n) {
    pack_bf16_k<<<1024, 256, 0, stream>>>(src, dst, n);
  };
  pack(query, qbf, ACT);  pack(key_, kbf, ACT);
  pack(value, vbf, ACT);  pack(iter, ibf, ACT);
  pack(Wq, wqbf, HIDN * HIDN);  pack(Wk, wkbf, HIDN * DD);
  pack(Wv, wvbf, HIDN * DD);    pack(Wiq, wiqbf, HIDN * HIDN);
  pack(Wik, wikbf, HIDN * DD);  pack(Wiv, wivbf, HIDN * DD);
  pack(Wo, wobf, HIDN * HIDN);

  // ---- projections (64x64 macro-tiles)
  auto gemm = [&](const __bf16* A, const __bf16* W, const float* bias,
                  float* Cf, __bf16* Cbf, int N, int transposeV) {
    int blocks = (M / 64) * (N / 64);
    wmma_gemm_bias<<<blocks, 256, 0, stream>>>(A, W, bias, Cf, Cbf,
                                               M, N, HIDN, transposeV);
  };
  gemm(qbf, wqbf,  bq,  nullptr, Qbf,  HIDN, 0);
  gemm(kbf, wkbf,  bk,  nullptr, Kbf,  DD,   0);
  gemm(vbf, wvbf,  bv,  nullptr, Vbf,  DD,   1);   // transposed
  gemm(ibf, wiqbf, biq, nullptr, Qibf, HIDN, 0);
  gemm(ibf, wikbf, bik, nullptr, Kibf, DD,   0);
  gemm(vbf, wivbf, biv, nullptr, Vibf, DD,   1);   // transposed

  // ---- fused attention (both paths) : 135168 B dynamic LDS
  const size_t lds = (size_t)(16 * SS + 16 * DD) * sizeof(float);
  hipFuncSetAttribute(reinterpret_cast<const void*>(attn_kernel),
                      hipFuncAttributeMaxDynamicSharedMemorySize, (int)lds);
  attn_kernel<<<dim3(SS / 16, HH, BB), 256, lds, stream>>>(
      Qbf, Qibf, Kbf, Kibf, Vbf, Vibf, mask, out_attn, Xbf);

  // ---- output projection -> d_out x region (f32)
  gemm(Xbf, wobf, bo, out_x, nullptr, HIDN, 0);
}
